// DeepGCN_60653528154451
// MI455X (gfx1250) — matrix operand order repaired and checked by reading, not compiled, not run on hardware
//
#include <hip/hip_runtime.h>
#include <hip/hip_bf16.h>
#include <math.h>

typedef __attribute__((ext_vector_type(16))) _Float16 v16h;
typedef __attribute__((ext_vector_type(8)))  _Float16 v8h;
typedef __attribute__((ext_vector_type(4)))  _Float16 v4h;
typedef __attribute__((ext_vector_type(8)))  float    v8f;

#define HDIM 256
#define CDIM 10
#define MT   5            // 16-row tiles per block
#define ROWS (MT * 16)    // 80 rows per block; 50000 / 80 = 625 blocks

__device__ __forceinline__ void atomic_add_f32(float* p, float v) {
  __hip_atomic_fetch_add(p, v, __ATOMIC_RELAXED, __HIP_MEMORY_SCOPE_AGENT);
}

// ---------------------------------------------------------------------------
// COO SpMM, H=256: out[rows[e], :] += w[e] * X[cols[e], :]
// one thread per (edge, float4-chunk); X and out are L2-resident (51.2MB < 192MB)
// ---------------------------------------------------------------------------
__global__ void spmm256_kernel(const int* __restrict__ rows, const int* __restrict__ cols,
                               const float* __restrict__ ew, const float* __restrict__ X,
                               float* __restrict__ out, int nE) {
  int t = blockIdx.x * blockDim.x + threadIdx.x;
  if (t >= nE * 64) return;
  int e = t >> 6;
  int q = (t & 63) << 2;
  int c = cols[e];
  int r = rows[e];
  float w = ew[e];
  const float4 x = *(const float4*)(X + (size_t)c * HDIM + q);
  float* o = out + (size_t)r * HDIM + q;
  atomic_add_f32(o + 0, w * x.x);
  atomic_add_f32(o + 1, w * x.y);
  atomic_add_f32(o + 2, w * x.z);
  atomic_add_f32(o + 3, w * x.w);
}

__global__ void relu_copy_kernel(const float* __restrict__ in, float* __restrict__ out, int n) {
  int t = blockIdx.x * blockDim.x + threadIdx.x;
  if (t < n) out[t] = fmaxf(in[t], 0.0f);
}

// ---------------------------------------------------------------------------
// Swizzle fp32 W[256][256] into this file's contiguous-k per-lane B-fragment order for
// V_WMMA_F32_16X16X32_F16: frag(kc, nt) lane L element j <-> W[kc*32 + 16*(L>>4) + j][nt*16 + (L&15)]
// Destination index == thread id with bit layout kc[15:13] nt[12:9] lane[8:4] j[3:0]
// ---------------------------------------------------------------------------
__global__ void conv_w_kernel(const float* __restrict__ W, _Float16* __restrict__ Bsw) {
  int t = blockIdx.x * blockDim.x + threadIdx.x;   // 0 .. 65535
  int j    =  t        & 15;
  int lane = (t >> 4)  & 31;
  int nt   = (t >> 9)  & 15;
  int kc   = (t >> 13) & 7;
  int K = kc * 32 + ((lane >> 4) << 4) + j;
  int n = (nt << 4) + (lane & 15);
  Bsw[t] = (_Float16)W[K * HDIM + n];
}

// ---------------------------------------------------------------------------
// Fused GCNII layer: support = 0.9*hi + 0.1*anchor
//   out = relu( theta * (support @ W) + (1-theta) * support )
// 256 threads = 8 waves; block owns an 80-row stripe. support is converted to
// f16 ONCE during staging, stored in LDS already in A-fragment layout (two
// 8-half parts per (mt,kc,lane) -> conflict-free ds_load_b128 pairs).
// Each wave owns 2 N-tiles and sweeps 5 M-tiles: per K-chunk, one B-fragment
// pair feeds 10 consecutive WMMAs (5x B reuse vs previous version).
// fp32 residual is recomputed from L2-resident hi/anchor in the epilogue.
// ---------------------------------------------------------------------------
__global__ void layer_kernel(const float* __restrict__ hi, const float* __restrict__ anchor,
                             const _Float16* __restrict__ Bsw, float* __restrict__ out,
                             float theta, int nN) {
  __shared__ _Float16 aFrag[MT * 8 * 2 * 32 * 8];   // 40 KB
  const int row0 = blockIdx.x * ROWS;

  // stage: support -> f16 A fragments (each thread: 20 float4 column-chunks)
  for (int v = threadIdx.x; v < ROWS * 64; v += 256) {
    int r = v >> 6;               // 0..79
    int c = (v & 63) << 2;        // 0..252, step 4
    int row = row0 + r;
    if (row >= nN) row = nN - 1;  // clamped read (uniform trip count keeps EXEC full)
    const float4 a = *(const float4*)(hi     + (size_t)row * HDIM + c);
    const float4 b = *(const float4*)(anchor + (size_t)row * HDIM + c);
    v4h hv;
    hv[0] = (_Float16)(0.9f * a.x + 0.1f * b.x);
    hv[1] = (_Float16)(0.9f * a.y + 0.1f * b.y);
    hv[2] = (_Float16)(0.9f * a.z + 0.1f * b.z);
    hv[3] = (_Float16)(0.9f * a.w + 0.1f * b.w);
    // (r, K=c..c+3) -> A-fragment coords
    int mt = r >> 4, M = r & 15;
    int kc = c >> 5, rem = c & 31;
    int half_, j;
    if (rem < 16) { half_ = 0; j = rem; }
    else          { half_ = 1; j = rem - 16; }
    int part = j >> 3, jj = j & 7;
    int lane = half_ * 16 + M;
    *(v4h*)&aFrag[(((((mt * 8 + kc) * 2 + part) * 32) + lane) << 3) + jj] = hv;
  }
  __syncthreads();

  const int lane  = threadIdx.x & 31;
  const int wave  = threadIdx.x >> 5;
  const int half_ = lane >> 4;
  const int lm    = lane & 15;
  const int nt0   = wave * 2, nt1 = nt0 + 1;

  v8f acc[MT][2] = {};
#pragma unroll
  for (int kc = 0; kc < 8; ++kc) {
    v16h b0 = *(const v16h*)(Bsw + (((kc * 16 + nt0) * 32 + lane) << 4));
    v16h b1 = *(const v16h*)(Bsw + (((kc * 16 + nt1) * 32 + lane) << 4));
#pragma unroll
    for (int mt = 0; mt < MT; ++mt) {
      v8h alo = *(const v8h*)&aFrag[((((mt * 8 + kc) * 2 + 0) * 32) + lane) << 3];
      v8h ahi = *(const v8h*)&aFrag[((((mt * 8 + kc) * 2 + 1) * 32) + lane) << 3];
      v16h a = __builtin_shufflevector(alo, ahi, 0, 1, 2, 3, 4, 5, 6, 7,
                                       8, 9, 10, 11, 12, 13, 14, 15);
      acc[mt][0] = __builtin_amdgcn_wmma_f32_16x16x32_f16(false, a, false, b0,
                                                          (short)0, acc[mt][0], false, false);
      acc[mt][1] = __builtin_amdgcn_wmma_f32_16x16x32_f16(false, a, false, b1,
                                                          (short)0, acc[mt][1], false, false);
    }
  }

  // epilogue: residual recomputed from L2-resident hi/anchor (exact fp32)
  const float omt = 1.0f - theta;
  const int c0 = nt0 * 16 + lm;
  const int c1 = nt1 * 16 + lm;
#pragma unroll
  for (int mt = 0; mt < MT; ++mt) {
#pragma unroll
    for (int r = 0; r < 8; ++r) {
      int row = row0 + mt * 16 + r + half_ * 8;
      if (row < nN) {
        size_t base = (size_t)row * HDIM;
        float s0 = 0.9f * hi[base + c0] + 0.1f * anchor[base + c0];
        float s1 = 0.9f * hi[base + c1] + 0.1f * anchor[base + c1];
        out[base + c0] = fmaxf(theta * acc[mt][0][r] + omt * s0, 0.0f);
        out[base + c1] = fmaxf(theta * acc[mt][1][r] + omt * s1, 0.0f);
      }
    }
  }
}

// h[N,256] @ W2[256,10] -> out[N,10]  (C=10 too narrow for WMMA; VALU dot)
__global__ void gemm_w2_kernel(const float* __restrict__ h, const float* __restrict__ W2,
                               float* __restrict__ out, int nN) {
  int t = blockIdx.x * blockDim.x + threadIdx.x;
  if (t >= nN * CDIM) return;
  int n = t / CDIM;
  int c = t % CDIM;
  const float* hr = h + (size_t)n * HDIM;
  float acc = 0.0f;
#pragma unroll 8
  for (int k = 0; k < HDIM; ++k) acc += hr[k] * W2[k * CDIM + c];
  out[t] = acc;
}

// final COO SpMM over C=10 columns
__global__ void spmmC_kernel(const int* __restrict__ rows, const int* __restrict__ cols,
                             const float* __restrict__ ew, const float* __restrict__ X,
                             float* __restrict__ out, int nE) {
  int t = blockIdx.x * blockDim.x + threadIdx.x;
  if (t >= nE * 16) return;
  int e = t >> 4;
  int q = t & 15;
  if (q >= CDIM) return;
  atomic_add_f32(out + (size_t)rows[e] * CDIM + q, ew[e] * X[(size_t)cols[e] * CDIM + q]);
}

extern "C" void kernel_launch(void* const* d_in, const int* in_sizes, int n_in,
                              void* d_out, int out_size, void* d_ws, size_t ws_size,
                              hipStream_t stream) {
  // setup_inputs order: x, rows, cols, edge_weight, W1, Wc, W2
  const int*   rows = (const int*)d_in[1];
  const int*   cols = (const int*)d_in[2];
  const float* ew   = (const float*)d_in[3];
  const float* W1   = (const float*)d_in[4];
  const float* Wc   = (const float*)d_in[5];
  const float* W2   = (const float*)d_in[6];
  const int N = in_sizes[0];   // 50000
  const int E = in_sizes[1];   // 1.6M
  float* out = (float*)d_out;

  // workspace layout (all offsets 32B-aligned)
  float* L0  = (float*)d_ws;                 // layer-1 output (== layer0 anchor)
  float* X1  = L0  + (size_t)N * HDIM;       // ping
  float* X2  = X1  + (size_t)N * HDIM;       // pong
  float* HI  = X2  + (size_t)N * HDIM;       // spmm accumulator
  float* HW2 = HI  + (size_t)N * HDIM;       // h @ W2
  _Float16* Bsw = (_Float16*)(HW2 + (size_t)N * CDIM);  // swizzled f16 W (current layer)

  const dim3 blk(256);
  const int spmm_blocks  = (E * 64 + 255) / 256;
  const int layer_blocks = (N + ROWS - 1) / ROWS;   // 625 for N=50000
  const size_t featBytes = (size_t)N * HDIM * sizeof(float);
  const int nNH = N * HDIM;

  // layer1: h = relu(spmm(W1))
  hipMemsetAsync(HI, 0, featBytes, stream);
  spmm256_kernel<<<spmm_blocks, blk, 0, stream>>>(rows, cols, ew, W1, HI, E);
  relu_copy_kernel<<<(nNH + 255) / 256, blk, 0, stream>>>(HI, L0, nNH);

  const float* h  = L0;
  const float* h0 = L0;
  float* pp[2] = {X1, X2};
  for (int i = 0; i < 8; ++i) {
    const float theta = logf(0.1f / (float)(i + 1) + 1.0f);
    conv_w_kernel<<<65536 / 256, blk, 0, stream>>>(Wc + (size_t)i * HDIM * HDIM, Bsw);
    hipMemsetAsync(HI, 0, featBytes, stream);
    spmm256_kernel<<<spmm_blocks, blk, 0, stream>>>(rows, cols, ew, h, HI, E);
    const float* anchor = (i <= 4) ? h0 : L0;
    float* o = pp[i & 1];
    layer_kernel<<<layer_blocks, blk, 0, stream>>>(HI, anchor, Bsw, o, theta, N);
    h = o;
    if (i <= 4) h0 = o;
  }

  // layer2: spmm(h @ W2)
  gemm_w2_kernel<<<(N * CDIM + 255) / 256, blk, 0, stream>>>(h, W2, HW2, N);
  hipMemsetAsync(out, 0, (size_t)N * CDIM * sizeof(float), stream);
  spmmC_kernel<<<(E * 16 + 255) / 256, blk, 0, stream>>>(rows, cols, ew, HW2, out, E);
}
